// GNS_48644799595230
// MI455X (gfx1250) — matrix operand order repaired and checked by reading, not compiled, not run on hardware
//
#include <hip/hip_runtime.h>
#include <hip/hip_bf16.h>
#include <stdint.h>

// =====================================================================
// GNS forward pass for MI455X (gfx1250, wave32, WMMA f16->f32).
//
// v2: each wave now computes TWO 16-row M-tiles (32 rows) so every
// B fragment pulled from LDS feeds two independent v_wmma ops
// (1 ds_load_b128 per WMMA instead of 2) — halves the LDS round trips
// behind s_wait_dscnt and doubles per-wave ILP for the scheduler.
//
// All 128-wide GEMMs run on v_wmma_f32_16x16x32_f16. Weights are packed
// once per launch into the exact WMMA B-fragment layout and staged in
// LDS (K<=384 -> <=96KB of the 320KB WGP LDS). The edge-MLP concat
// [edges | nodes[r] | nodes[s]] (K=384) and node-MLP concat
// [agg_f32 | nodes] (K=256) are fused into the A-fragment loader.
// bias+ReLU and bias+LayerNorm+residual are fused into the epilogue;
// LN stats via half-wave shfl_xor (WMMA C layout keeps a row in 16
// lanes). Residual streams f32; GEMMs consume f16 shadows. segment_sum
// via f32 L2 atomics.
// =====================================================================

typedef __attribute__((ext_vector_type(16))) _Float16 v16h;
typedef __attribute__((ext_vector_type(8)))  float    v8f;

#define EMB 128
#define LNEPS 1e-5f
#define RWALL 0.015f

struct GemmArgs {
  const _Float16* A0;    // mode0: X ; mode1: edges_f16 ; mode2: nodes_f16 (seg1)
  const _Float16* A1;    // mode1: nodes_f16 (gathered via r/s)
  const float*    Af;    // mode2: agg f32 (seg0)
  const int*      nidx;  // neighbor_idxs [E,3] (d,r,s)
  const _Float16* Wt;    // pre-tiled WMMA-B weights, KT*4096 halves
  const float* bias;
  const float* g;
  const float* bv;
  const float* resid;    // nullable (encoders)
  float*       outF;
  _Float16*    outH;
  int M;
  int strideA;           // mode0 only (elements)
};

union AFrag { v16h v; uint4 q[2]; _Float16 h[16]; };
union BFrag { v16h v; uint4 q[2]; };

// ---------------------------------------------------------------------
// Fused MLP-layer GEMM: Y[M,128] = epi( A[M,32*KT] @ W + bias )
// Block = 256 threads = 8 waves; block tile = 256 rows x 128 cols;
// wave w owns rows [w*32, w*32+32) (two 16-row M-tiles) x 8 col tiles.
// ---------------------------------------------------------------------
template<int KT, int MODE, bool RELU, bool LNEPI>
__global__ __launch_bounds__(256)
void gemm_mlp(GemmArgs ga)
{
  extern __shared__ uint4 smW[];                 // KT*512 uint4 (KT*8KB)
  {
    const uint4* wsrc = (const uint4*)ga.Wt;
    for (int i = threadIdx.x; i < KT * 512; i += 256) smW[i] = wsrc[i];
  }
  __syncthreads();

  const int lane  = threadIdx.x & 31;
  const int wave  = threadIdx.x >> 5;
  const int laneM = lane & 15;
  const int kHalf = (lane >> 4) * 8;             // A layout: hi half-wave holds K+8
  const int rowBase = blockIdx.x * 256 + wave * 32;

  int row[2];
  row[0] = rowBase + laneM;
  row[1] = rowBase + 16 + laneM;
  #pragma unroll
  for (int u = 0; u < 2; ++u)
    if (row[u] >= ga.M) row[u] = ga.M - 1;       // clamp: EXEC stays all-ones for WMMA

  const _Float16 *p0[2] = {nullptr, nullptr};
  const _Float16 *p1[2] = {nullptr, nullptr};
  const _Float16 *p2[2] = {nullptr, nullptr};
  const float    *pf[2] = {nullptr, nullptr};
  #pragma unroll
  for (int u = 0; u < 2; ++u) {
    if (MODE == 0) {
      p0[u] = ga.A0 + (size_t)row[u] * ga.strideA;
    } else if (MODE == 1) {
      p0[u] = ga.A0 + (size_t)row[u] * EMB;
      const int r = ga.nidx[row[u] * 3 + 1];
      const int s = ga.nidx[row[u] * 3 + 2];
      p1[u] = ga.A1 + (size_t)r * EMB;
      p2[u] = ga.A1 + (size_t)s * EMB;
    } else {
      pf[u] = ga.Af + (size_t)row[u] * EMB;
      p0[u] = ga.A0 + (size_t)row[u] * EMB;
    }
  }

  v8f acc[2][8] = {};

  #pragma unroll
  for (int kt = 0; kt < KT; ++kt) {
    AFrag a[2];
    #pragma unroll
    for (int u = 0; u < 2; ++u) {
      if (MODE == 0) {
        const _Float16* p = p0[u] + kt * 32 + kHalf;
        a[u].q[0] = *(const uint4*)(p);
        a[u].q[1] = *(const uint4*)(p + 16);
      } else if (MODE == 1) {
        const int seg = kt >> 2;                 // 128-wide concat segments
        const _Float16* pb = (seg == 0) ? p0[u] : ((seg == 1) ? p1[u] : p2[u]);
        const _Float16* p = pb + (kt & 3) * 32 + kHalf;
        a[u].q[0] = *(const uint4*)(p);
        a[u].q[1] = *(const uint4*)(p + 16);
      } else {
        if (kt < 4) {                            // agg segment is f32: convert on load
          const float* p = pf[u] + kt * 32 + kHalf;
          float4 x0 = ((const float4*)p)[0];
          float4 x1 = ((const float4*)p)[1];
          float4 x2 = ((const float4*)(p + 16))[0];
          float4 x3 = ((const float4*)(p + 16))[1];
          a[u].h[0]=(_Float16)x0.x;  a[u].h[1]=(_Float16)x0.y;  a[u].h[2]=(_Float16)x0.z;  a[u].h[3]=(_Float16)x0.w;
          a[u].h[4]=(_Float16)x1.x;  a[u].h[5]=(_Float16)x1.y;  a[u].h[6]=(_Float16)x1.z;  a[u].h[7]=(_Float16)x1.w;
          a[u].h[8]=(_Float16)x2.x;  a[u].h[9]=(_Float16)x2.y;  a[u].h[10]=(_Float16)x2.z; a[u].h[11]=(_Float16)x2.w;
          a[u].h[12]=(_Float16)x3.x; a[u].h[13]=(_Float16)x3.y; a[u].h[14]=(_Float16)x3.z; a[u].h[15]=(_Float16)x3.w;
        } else {
          const _Float16* p = p0[u] + (kt - 4) * 32 + kHalf;
          a[u].q[0] = *(const uint4*)(p);
          a[u].q[1] = *(const uint4*)(p + 16);
        }
      }
    }
    #pragma unroll
    for (int t = 0; t < 8; ++t) {
      BFrag b;                                   // one ds_load_b128 pair feeds TWO wmmas
      const int bi = ((kt * 8 + t) * 32 + lane) * 2;
      b.q[0] = smW[bi];
      b.q[1] = smW[bi + 1];
      acc[0][t] = __builtin_amdgcn_wmma_f32_16x16x32_f16(
          false, a[0].v, false, b.v, (short)0, acc[0][t], false, false);
      acc[1][t] = __builtin_amdgcn_wmma_f32_16x16x32_f16(
          false, a[1].v, false, b.v, (short)0, acc[1][t], false, false);
    }
  }

  #pragma unroll
  for (int u = 0; u < 2; ++u) {
    const int mBase = rowBase + u * 16 + (lane >> 4) * 8;  // C layout: VGPR j -> row mBase+j

    if (!LNEPI) {
      #pragma unroll
      for (int t = 0; t < 8; ++t) {
        const int col = t * 16 + laneM;
        const float bb = ga.bias[col];
        #pragma unroll
        for (int j = 0; j < 8; ++j) {
          float v = acc[u][t][j] + bb;
          if (RELU) v = fmaxf(v, 0.0f);
          const int m = mBase + j;
          if (m < ga.M) ga.outH[(size_t)m * EMB + col] = (_Float16)v;
        }
      }
    } else {
      // bias in place, then per-row LN stats across the 16 lanes holding the row
      #pragma unroll
      for (int t = 0; t < 8; ++t) {
        const float bb = ga.bias[t * 16 + laneM];
        #pragma unroll
        for (int j = 0; j < 8; ++j) acc[u][t][j] += bb;
      }
      #pragma unroll
      for (int j = 0; j < 8; ++j) {
        float s = 0.f, s2 = 0.f;
        #pragma unroll
        for (int t = 0; t < 8; ++t) { const float v = acc[u][t][j]; s += v; s2 += v * v; }
        #pragma unroll
        for (int xm = 1; xm < 16; xm <<= 1) {    // xor<16 keeps lanes inside their half
          s  += __shfl_xor(s,  xm, 32);
          s2 += __shfl_xor(s2, xm, 32);
        }
        const float mean = s * (1.0f / 128.0f);
        const float var  = s2 * (1.0f / 128.0f) - mean * mean;
        const float rstd = rsqrtf(var + LNEPS);
        const int m = mBase + j;
        if (m < ga.M) {
          #pragma unroll
          for (int t = 0; t < 8; ++t) {
            const int col = t * 16 + laneM;
            const size_t idx = (size_t)m * EMB + col;
            const float y = (acc[u][t][j] - mean) * rstd * ga.g[col] + ga.bv[col];
            const float r = ga.resid ? ga.resid[idx] : 0.0f;
            const float o = r + y;
            ga.outF[idx] = o;                    // f32 residual stream
            ga.outH[idx] = (_Float16)o;          // f16 shadow for next GEMM
          }
        }
      }
    }
  }
}

// ---------------------------------------------------------------------
// Pack W[Kreal,128] (f32, row-major, x@W convention) into the WMMA-B
// fragment layout, zero-padded to KP rows:
//   out[((kt*8+t)*32+lane)*16+j] = W[kt*32 + (lane>>4)*8 + (j&7) + (j>>3)*16][t*16 + (lane&15)]
// ---------------------------------------------------------------------
__global__ void pack_w_k(const float* __restrict__ W, _Float16* __restrict__ out,
                         int Kreal, int KP)
{
  const int id = blockIdx.x * 256 + threadIdx.x;
  if (id >= KP * 128) return;
  const int j    = id & 15;
  const int lane = (id >> 4) & 31;
  const int t    = (id >> 9) & 7;
  const int kt   = id >> 12;
  const int k    = kt * 32 + (lane >> 4) * 8 + (j & 7) + (j >> 3) * 16;
  const int col  = t * 16 + (lane & 15);
  const float v = (k < Kreal) ? W[k * 128 + col] : 0.0f;
  out[id] = (_Float16)v;
}

__global__ void fill0_k(float* p, int n)
{
  const int i = blockIdx.x * 256 + threadIdx.x;
  if (i < n) p[i] = 0.0f;
}

// node features (padded to K=64): [vel(15) | matW[m]+matb(16) | wall(6) | 0...]
__global__ void node_feats_k(const float* __restrict__ pos, const float* __restrict__ vel,
                             const int* __restrict__ mat, const float* __restrict__ matW,
                             const float* __restrict__ matB, _Float16* __restrict__ out, int N)
{
  const int n = blockIdx.x * 256 + threadIdx.x;
  if (n >= N) return;
  union { _Float16 h[64]; uint4 q[8]; } f;
  #pragma unroll
  for (int i = 0; i < 64; ++i) f.h[i] = (_Float16)0.0f;
  #pragma unroll
  for (int i = 0; i < 15; ++i) f.h[i] = (_Float16)vel[n * 15 + i];
  const int m = mat[n];
  #pragma unroll
  for (int j = 0; j < 16; ++j) f.h[15 + j] = (_Float16)(matW[m * 16 + j] + matB[j]);
  #pragma unroll
  for (int d = 0; d < 3; ++d) {
    const float p = pos[n * 3 + d];
    f.h[31 + 2 * d]     = (_Float16)fminf(fmaxf(p,        0.0f), RWALL);
    f.h[31 + 2 * d + 1] = (_Float16)fminf(fmaxf(1.0f - p, 0.0f), RWALL);
  }
  uint4* o = (uint4*)(out + (size_t)n * 64);
  #pragma unroll
  for (int i = 0; i < 8; ++i) o[i] = f.q[i];
}

// edge features (padded to K=32): [rel(3) | dist(1) | 0...]
__global__ void edge_feats_k(const float* __restrict__ pos, const int* __restrict__ ni,
                             _Float16* __restrict__ out, int E)
{
  const int e = blockIdx.x * 256 + threadIdx.x;
  if (e >= E) return;
  const int r = ni[e * 3 + 1];
  const int s = ni[e * 3 + 2];
  const float dx = pos[r * 3 + 0] - pos[s * 3 + 0];
  const float dy = pos[r * 3 + 1] - pos[s * 3 + 1];
  const float dz = pos[r * 3 + 2] - pos[s * 3 + 2];
  union { _Float16 h[32]; uint4 q[4]; } f;
  #pragma unroll
  for (int i = 0; i < 32; ++i) f.h[i] = (_Float16)0.0f;
  f.h[0] = (_Float16)dx; f.h[1] = (_Float16)dy; f.h[2] = (_Float16)dz;
  f.h[3] = (_Float16)sqrtf(dx * dx + dy * dy + dz * dz);
  uint4* o = (uint4*)(out + (size_t)e * 32);
  #pragma unroll
  for (int i = 0; i < 4; ++i) o[i] = f.q[i];
}

// segment_sum over receivers: one thread per (edge, col), f32 L2 atomics
__global__ void scatter_k(const float* __restrict__ edgesF, const int* __restrict__ ni,
                          float* __restrict__ agg, int total)
{
  const int id = blockIdx.x * 256 + threadIdx.x;
  if (id >= total) return;
  const int e = id >> 7, k = id & 127;
  const int r = ni[e * 3 + 1];
  atomicAdd(&agg[(size_t)r * EMB + k], edgesF[id]);
}

// decoder tail: 128 -> 3 linear (tiny, VALU); vectorized h2 loads
__global__ void dec_final_k(const _Float16* __restrict__ h2, const float* __restrict__ W2,
                            const float* __restrict__ b2, float* __restrict__ out, int N)
{
  const int n = blockIdx.x * 256 + threadIdx.x;
  if (n >= N) return;
  float a0 = b2[0], a1 = b2[1], a2 = b2[2];
  const uint4* hv = (const uint4*)(h2 + (size_t)n * 128);
  #pragma unroll
  for (int i = 0; i < 16; ++i) {
    union { uint4 q; _Float16 h[8]; } u;
    u.q = hv[i];
    #pragma unroll
    for (int j = 0; j < 8; ++j) {
      const int k = i * 8 + j;
      const float h = (float)u.h[j];
      a0 += h * W2[k * 3 + 0];
      a1 += h * W2[k * 3 + 1];
      a2 += h * W2[k * 3 + 2];
    }
  }
  out[n * 3 + 0] = a0; out[n * 3 + 1] = a1; out[n * 3 + 2] = a2;
}

// =====================================================================
#define GEMM(KT, MODE, RELU, LNE, A) \
  gemm_mlp<KT, MODE, RELU, LNE><<<dim3(((A).M + 255) / 256), 256, (KT) * 8192, stream>>>(A)

extern "C" void kernel_launch(void* const* d_in, const int* in_sizes, int n_in,
                              void* d_out, int out_size, void* d_ws, size_t ws_size,
                              hipStream_t stream)
{
  (void)n_in; (void)out_size; (void)ws_size;

  // --- flatten inputs in setup_inputs() insertion order ---------------
  int ix = 0;
  const float* positions  = (const float*)d_in[ix++];
  const float* velocities = (const float*)d_in[ix++];
  const float* mat_W      = (const float*)d_in[ix++];
  const float* mat_b      = (const float*)d_in[ix++];

  struct MLPp { const float *W0, *W1, *W2, *b0, *b1, *b2; };
  auto takeMLP = [&]() {
    MLPp m;
    m.W0 = (const float*)d_in[ix++]; m.W1 = (const float*)d_in[ix++];
    m.W2 = (const float*)d_in[ix++];
    m.b0 = (const float*)d_in[ix++]; m.b1 = (const float*)d_in[ix++];
    m.b2 = (const float*)d_in[ix++];
    return m;
  };

  MLPp neM = takeMLP();
  const float* ne_g = (const float*)d_in[ix++];
  const float* ne_b = (const float*)d_in[ix++];
  MLPp eeM = takeMLP();
  const float* ee_g = (const float*)d_in[ix++];
  const float* ee_b = (const float*)d_in[ix++];

  struct ProcP { MLPp eM; const float *eg, *eb; MLPp nM; const float *ng, *nb; };
  ProcP P[10];
  for (int l = 0; l < 10; ++l) {
    P[l].eM = takeMLP();
    P[l].eg = (const float*)d_in[ix++];
    P[l].eb = (const float*)d_in[ix++];
    P[l].nM = takeMLP();
    P[l].ng = (const float*)d_in[ix++];
    P[l].nb = (const float*)d_in[ix++];
  }
  MLPp dec = takeMLP();
  const int* materials = (const int*)d_in[ix++];
  const int* nidx      = (const int*)d_in[ix++];

  const int N = in_sizes[0] / 3;
  const int E = in_sizes[ix - 1] / 3;

  // --- workspace carve-out --------------------------------------------
  char* wsBase = (char*)d_ws; size_t wsOff = 0;
  auto alloc = [&](size_t bytes) -> void* {
    void* p = wsBase + wsOff;
    wsOff += (bytes + 255) & ~(size_t)255;
    return p;
  };
  auto packW = [&](const float* src, int Kreal, int KP) -> const _Float16* {
    _Float16* dst = (_Float16*)alloc((size_t)KP * 128 * sizeof(_Float16));
    const int total = KP * 128;
    pack_w_k<<<dim3((total + 255) / 256), 256, 0, stream>>>(src, dst, Kreal, KP);
    return dst;
  };

  // pack all GEMM weights into WMMA-B tiled f16 (per-launch; no caching)
  const _Float16* ne_tW0 = packW(neM.W0, 37, 64);
  const _Float16* ne_tW1 = packW(neM.W1, 128, 128);
  const _Float16* ne_tW2 = packW(neM.W2, 128, 128);
  const _Float16* ee_tW0 = packW(eeM.W0, 4, 32);
  const _Float16* ee_tW1 = packW(eeM.W1, 128, 128);
  const _Float16* ee_tW2 = packW(eeM.W2, 128, 128);
  const _Float16 *pe_tW0[10], *pe_tW1[10], *pe_tW2[10];
  const _Float16 *pn_tW0[10], *pn_tW1[10], *pn_tW2[10];
  for (int l = 0; l < 10; ++l) {
    pe_tW0[l] = packW(P[l].eM.W0, 384, 384);
    pe_tW1[l] = packW(P[l].eM.W1, 128, 128);
    pe_tW2[l] = packW(P[l].eM.W2, 128, 128);
    pn_tW0[l] = packW(P[l].nM.W0, 256, 256);
    pn_tW1[l] = packW(P[l].nM.W1, 128, 128);
    pn_tW2[l] = packW(P[l].nM.W2, 128, 128);
  }
  const _Float16* dc_tW0 = packW(dec.W0, 128, 128);
  const _Float16* dc_tW1 = packW(dec.W1, 128, 128);

  // activations
  _Float16* nodeFeat = (_Float16*)alloc((size_t)N * 64 * 2);
  _Float16* edgeFeat = (_Float16*)alloc((size_t)E * 32 * 2);
  _Float16* nodesH   = (_Float16*)alloc((size_t)N * 128 * 2);
  float*    nodesF   = (float*)   alloc((size_t)N * 128 * 4);
  _Float16* edgesH   = (_Float16*)alloc((size_t)E * 128 * 2);
  float*    edgesF   = (float*)   alloc((size_t)E * 128 * 4);
  float*    agg      = (float*)   alloc((size_t)N * 128 * 4);
  _Float16* h1       = (_Float16*)alloc((size_t)E * 128 * 2);
  _Float16* h2       = (_Float16*)alloc((size_t)E * 128 * 2);

  auto GA = [&](const _Float16* A0, int strideA, const _Float16* Wt,
                const float* bias, int M) {
    GemmArgs a{};
    a.A0 = A0; a.strideA = strideA; a.Wt = Wt; a.bias = bias; a.M = M;
    a.nidx = nidx;
    return a;
  };

  // --- feature builders -----------------------------------------------
  node_feats_k<<<dim3((N + 255) / 256), 256, 0, stream>>>(
      positions, velocities, materials, mat_W, mat_b, nodeFeat, N);
  edge_feats_k<<<dim3((E + 255) / 256), 256, 0, stream>>>(positions, nidx, edgeFeat, E);

  // --- node encoder (37->128->128->128, LN) ---------------------------
  { GemmArgs a = GA(nodeFeat, 64, ne_tW0, neM.b0, N); a.outH = h1; GEMM(2, 0, true, false, a); }
  { GemmArgs a = GA(h1, 128, ne_tW1, neM.b1, N); a.outH = h2; GEMM(4, 0, true, false, a); }
  { GemmArgs a = GA(h2, 128, ne_tW2, neM.b2, N);
    a.g = ne_g; a.bv = ne_b; a.resid = nullptr; a.outF = nodesF; a.outH = nodesH;
    GEMM(4, 0, false, true, a); }

  // --- edge encoder (4->128->128->128, LN) ----------------------------
  { GemmArgs a = GA(edgeFeat, 32, ee_tW0, eeM.b0, E); a.outH = h1; GEMM(1, 0, true, false, a); }
  { GemmArgs a = GA(h1, 128, ee_tW1, eeM.b1, E); a.outH = h2; GEMM(4, 0, true, false, a); }
  { GemmArgs a = GA(h2, 128, ee_tW2, eeM.b2, E);
    a.g = ee_g; a.bv = ee_b; a.resid = nullptr; a.outF = edgesF; a.outH = edgesH;
    GEMM(4, 0, false, true, a); }

  // --- 10 message-passing layers --------------------------------------
  for (int l = 0; l < 10; ++l) {
    // edge MLP: fused gather-concat [edges|nodes[r]|nodes[s]] (K=384)
    { GemmArgs a = GA(edgesH, 128, pe_tW0[l], P[l].eM.b0, E);
      a.A1 = nodesH; a.outH = h1; GEMM(12, 1, true, false, a); }
    { GemmArgs a = GA(h1, 128, pe_tW1[l], P[l].eM.b1, E); a.outH = h2; GEMM(4, 0, true, false, a); }
    { GemmArgs a = GA(h2, 128, pe_tW2[l], P[l].eM.b2, E);
      a.g = P[l].eg; a.bv = P[l].eb; a.resid = edgesF; a.outF = edgesF; a.outH = edgesH;
      GEMM(4, 0, false, true, a); }

    // segment_sum(edges) over receivers
    fill0_k<<<dim3((N * 128 + 255) / 256), 256, 0, stream>>>(agg, N * 128);
    scatter_k<<<dim3((E * 128 + 255) / 256), 256, 0, stream>>>(edgesF, nidx, agg, E * 128);

    // node MLP: fused concat [agg(f32)|nodes] (K=256)
    { GemmArgs a = GA(nodesH, 128, pn_tW0[l], P[l].nM.b0, N);
      a.Af = agg; a.outH = h1; GEMM(8, 2, true, false, a); }
    { GemmArgs a = GA(h1, 128, pn_tW1[l], P[l].nM.b1, N); a.outH = h2; GEMM(4, 0, true, false, a); }
    { GemmArgs a = GA(h2, 128, pn_tW2[l], P[l].nM.b2, N);
      a.g = P[l].ng; a.bv = P[l].nb; a.resid = nodesF; a.outF = nodesF; a.outH = nodesH;
      GEMM(4, 0, false, true, a); }
  }

  // --- decoder (128->128->128->3) -------------------------------------
  { GemmArgs a = GA(nodesH, 128, dc_tW0, dec.b0, N); a.outH = h1; GEMM(4, 0, true, false, a); }
  { GemmArgs a = GA(h1, 128, dc_tW1, dec.b1, N); a.outH = h2; GEMM(4, 0, true, false, a); }
  dec_final_k<<<dim3((N + 255) / 256), 256, 0, stream>>>(h2, dec.W2, dec.b2, (float*)d_out, N);
}
